// DiffusionTransformerBlock_61203283968333
// MI455X (gfx1250) — matrix operand order repaired and checked by reading, not compiled, not run on hardware
//
#include <hip/hip_runtime.h>
#include <hip/hip_bf16.h>
#include <math.h>

// ---------------------------------------------------------------------------
// DiffusionTransformerBlock (AF3 Alg 24+25) for MI455X / gfx1250.
// - All big GEMMs: v_wmma_f32_16x16x32_f16 (f16 in, f32 accumulate).
// - Weights pre-converted once to f16 transposed [N,K] so GEMM staging is
//   pure 16B copies via global_load_async_to_lds_b128 (double-buffered).
// - pair_rep (512 MB) read exactly once -> HBM-bound, ~25us floor @23.3TB/s.
// Workspace: ~135 MB from d_ws (bump allocator below).
// ---------------------------------------------------------------------------

typedef __attribute__((ext_vector_type(16))) _Float16 v16h;
typedef __attribute__((ext_vector_type(8)))  _Float16 v8h;
typedef __attribute__((ext_vector_type(8)))  float    v8f;

#define N_TOK   1024
#define C_DIM   768
#define CP_DIM  128
#define H_NUM   16
#define D_DIM   48
#define HID_DIM 1536

__device__ __forceinline__ float sig_(float x) { return 1.0f / (1.0f + __expf(-x)); }

__device__ __forceinline__ v8f wmma_f16(v16h a, v16h b, v8f c) {
  // (neg_a, A, neg_b, B, c_mod, C, reuse_a, reuse_b)
  return __builtin_amdgcn_wmma_f32_16x16x32_f16(false, a, false, b, (short)0, c,
                                                false, false);
}

// CDNA5 async memory->LDS copy (16B per lane), tracked by ASYNCcnt.
__device__ __forceinline__ void async_copy16(uint32_t lds_addr, const void* gaddr) {
  asm volatile("global_load_async_to_lds_b128 %0, %1, off"
               :: "v"(lds_addr), "v"((uint64_t)(uintptr_t)gaddr)
               : "memory");
}
__device__ __forceinline__ void wait_async0() {
  asm volatile("s_wait_asynccnt 0" ::: "memory");
}

// ---------------------------------------------------------------------------
// 0) Weight pre-conversion: W f32 [K,Nc] -> Wt f16 [Nc,K] (32x32 LDS tiles).
// ---------------------------------------------------------------------------
__global__ __launch_bounds__(256)
void convert_w_kernel(const float* __restrict__ W, _Float16* __restrict__ Wt,
                      int K, int Nc) {
  __shared__ _Float16 t[32][33];
  const int tx = threadIdx.x & 31, ty = threadIdx.x >> 5;  // 32x8, 4 passes
  const int bc = blockIdx.x * 32, bk = blockIdx.y * 32;
#pragma unroll
  for (int p = 0; p < 4; ++p) {
    int r = bk + ty + p * 8, c = bc + tx;
    t[ty + p * 8][tx] = (_Float16)W[(size_t)r * Nc + c];
  }
  __syncthreads();
#pragma unroll
  for (int p = 0; p < 4; ++p) {
    int c = bc + ty + p * 8, r = bk + tx;
    Wt[(size_t)c * K + r] = t[tx][ty + p * 8];
  }
}

// ---------------------------------------------------------------------------
// 1) Row-LN prep: an_raw = LN(a); s_ln = LN(s); f16 copies of s_ln*sg{1,2}, s.
// ---------------------------------------------------------------------------
__global__ __launch_bounds__(256)
void ln_prep_kernel(const float* __restrict__ a, const float* __restrict__ s,
                    const float* __restrict__ sg1, const float* __restrict__ sg2,
                    float* __restrict__ an_raw, _Float16* __restrict__ sn1h,
                    _Float16* __restrict__ sn2h, _Float16* __restrict__ sh_out) {
  __shared__ float redA[256], redS[256];
  const int row = blockIdx.x, tid = threadIdx.x;
  const float* ar = a + (size_t)row * C_DIM;
  const float* sr = s + (size_t)row * C_DIM;
  float va[3], vs[3], sa = 0.f, ss = 0.f;
#pragma unroll
  for (int i = 0; i < 3; ++i) {
    int c = tid + 256 * i;
    va[i] = ar[c]; vs[i] = sr[c];
    sa += va[i]; ss += vs[i];
  }
  redA[tid] = sa; redS[tid] = ss; __syncthreads();
  for (int off = 128; off > 0; off >>= 1) {
    if (tid < off) { redA[tid] += redA[tid + off]; redS[tid] += redS[tid + off]; }
    __syncthreads();
  }
  const float muA = redA[0] * (1.0f / C_DIM);
  const float muS = redS[0] * (1.0f / C_DIM);
  __syncthreads();
  sa = 0.f; ss = 0.f;
#pragma unroll
  for (int i = 0; i < 3; ++i) {
    float da = va[i] - muA, ds = vs[i] - muS;
    sa += da * da; ss += ds * ds;
  }
  redA[tid] = sa; redS[tid] = ss; __syncthreads();
  for (int off = 128; off > 0; off >>= 1) {
    if (tid < off) { redA[tid] += redA[tid + off]; redS[tid] += redS[tid + off]; }
    __syncthreads();
  }
  const float rsA = rsqrtf(redA[0] * (1.0f / C_DIM) + 1e-5f);
  const float rsS = rsqrtf(redS[0] * (1.0f / C_DIM) + 1e-5f);
#pragma unroll
  for (int i = 0; i < 3; ++i) {
    int c = tid + 256 * i;
    size_t idx = (size_t)row * C_DIM + c;
    an_raw[idx] = (va[i] - muA) * rsA;
    float sl = (vs[i] - muS) * rsS;
    sn1h[idx] = (_Float16)(sl * sg1[c]);
    sn2h[idx] = (_Float16)(sl * sg2[c]);
    sh_out[idx] = (_Float16)vs[i];
  }
}

// ---------------------------------------------------------------------------
// 2) WMMA GEMM: out = act(scale*(A[MxK](f16) @ Wt[NcxK](f16,transposed)+bias))
//    64x64x64 block tile, 8 waves x two 16x16 C-tiles sharing one B-frag,
//    double-buffered LDS filled with global_load_async_to_lds_b128.
//    act: 0=none, 1=sigmoid, 2=silu.
// ---------------------------------------------------------------------------
#define BM 64
#define BN 64
#define BKK 64

__global__ __launch_bounds__(256)
void gemm_kernel(const _Float16* __restrict__ A, const _Float16* __restrict__ Wt,
                 const float* __restrict__ bias, float* __restrict__ outF,
                 _Float16* __restrict__ outH, int M, int K, int Nc, int act,
                 float scale) {
  __shared__ _Float16 lA[2][BM * BKK];  // [row][k]
  __shared__ _Float16 lB[2][BN * BKK];  // [col][k]
  const int tid = threadIdx.x;
  const int wave = tid >> 5, lane = tid & 31;
  const int laneRow = lane & 15, laneHi = lane >> 4;
  const int m0 = blockIdx.y * BM, n0 = blockIdx.x * BN;
  const int tr0 = wave >> 2, tc = wave & 3;  // C tiles (tr0,tc),(tr0+2,tc)
  v8f acc0 = {0.f, 0.f, 0.f, 0.f, 0.f, 0.f, 0.f, 0.f};
  v8f acc1 = acc0;

  // Staging: 64x64 halves = 512 x 16B chunks per tile; 2 chunks per thread.
  const int r0 = tid >> 3, c0 = (tid & 7) * 8;  // rows 0..31
  const int r1 = r0 + 32;                       // rows 32..63
  auto stage = [&](int buf, int k0) {
    const _Float16* Ab = A + (size_t)m0 * K + k0;
    const _Float16* Bb = Wt + (size_t)n0 * K + k0;
    async_copy16((uint32_t)(uintptr_t)&lA[buf][r0 * BKK + c0], Ab + (size_t)r0 * K + c0);
    async_copy16((uint32_t)(uintptr_t)&lA[buf][r1 * BKK + c0], Ab + (size_t)r1 * K + c0);
    async_copy16((uint32_t)(uintptr_t)&lB[buf][r0 * BKK + c0], Bb + (size_t)r0 * K + c0);
    async_copy16((uint32_t)(uintptr_t)&lB[buf][r1 * BKK + c0], Bb + (size_t)r1 * K + c0);
  };

  stage(0, 0);
  const int nstage = K / BKK;
  for (int s = 0; s < nstage; ++s) {
    wait_async0();       // my async loads into buf(s) (and buf(s+1) issue slot)
    __syncthreads();     // everyone's loads into buf(s) are now visible
    if (s + 1 < nstage) stage((s + 1) & 1, (s + 1) * BKK);
    const _Float16* cA = &lA[s & 1][0];
    const _Float16* cB = &lB[s & 1][0];
#pragma unroll
    for (int ks = 0; ks < 2; ++ks) {
      const int base = ks * 32 + (laneHi ? 16 : 0);
      v16h af0, af1, bf;
      const v8h* p0 = (const v8h*)&cA[(tr0 * 16 + laneRow) * BKK + base];
      const v8h* p1 = (const v8h*)&cA[((tr0 + 2) * 16 + laneRow) * BKK + base];
      v8h lo0 = p0[0], hi0 = p0[1];
      v8h lo1 = p1[0], hi1 = p1[1];
#pragma unroll
      for (int e = 0; e < 8; ++e) {
        af0[e] = lo0[e]; af0[e + 8] = hi0[e];
        af1[e] = lo1[e]; af1[e + 8] = hi1[e];
      }
      bf = *(const v16h*)&cB[(tc * 16 + laneRow) * BKK + ks * 32 + (laneHi ? 16 : 0)];
      acc0 = wmma_f16(af0, bf, acc0);
      acc1 = wmma_f16(af1, bf, acc1);
    }
    __syncthreads();  // all reads of buf(s) done before it is refilled (s+2)
  }

  auto store = [&](v8f acc, int tr) {
    const int col = n0 + tc * 16 + laneRow;
    const float bv = bias ? bias[col] : 0.f;
#pragma unroll
    for (int e = 0; e < 8; ++e) {
      int m = m0 + tr * 16 + e + 8 * laneHi;
      float v = (acc[e] + bv) * scale;
      if (act == 1) v = sig_(v);
      else if (act == 2) v = v * sig_(v);
      if (outF) outF[(size_t)m * Nc + col] = v;
      if (outH) outH[(size_t)m * Nc + col] = (_Float16)v;
    }
  };
  store(acc0, tr0);
  store(acc1, tr0 + 2);
}

// ---------------------------------------------------------------------------
// 3) an = sigmoid(..)*LN(a) + skip, for both branches, f16 out.
// ---------------------------------------------------------------------------
__global__ __launch_bounds__(256)
void ada_combine_kernel(const float* __restrict__ g1, const float* __restrict__ sk1,
                        const float* __restrict__ g2, const float* __restrict__ sk2,
                        const float* __restrict__ an_raw,
                        _Float16* __restrict__ anh, _Float16* __restrict__ an2h) {
  size_t i = (size_t)blockIdx.x * 256 + threadIdx.x;
  float base = an_raw[i];
  anh[i]  = (_Float16)(g1[i] * base + sk1[i]);
  an2h[i] = (_Float16)(g2[i] * base + sk2[i]);
}

// ---------------------------------------------------------------------------
// 4) Pair bias, one pass over pair_rep (512MB):
//    biasT[h][i][j] = (LN(pair[i,j])*gp+bp) @ wp[:,h] + beta[i,j]
// ---------------------------------------------------------------------------
__global__ __launch_bounds__(128)
void pair_bias_kernel(const float* __restrict__ pair, const float* __restrict__ beta,
                      const float* __restrict__ gp, const float* __restrict__ bp,
                      const float* __restrict__ wp, float* __restrict__ biasT) {
  __shared__ float r[128];
  __shared__ float zn[128];
  const int j = blockIdx.x, i = blockIdx.y, tid = threadIdx.x;
  const size_t base = ((size_t)i * N_TOK + j) * CP_DIM;
  const float z = pair[base + tid];
  r[tid] = z; __syncthreads();
  for (int off = 64; off > 0; off >>= 1) {
    if (tid < off) r[tid] += r[tid + off];
    __syncthreads();
  }
  const float mu = r[0] * (1.0f / CP_DIM);
  __syncthreads();
  const float d = z - mu;
  r[tid] = d * d; __syncthreads();
  for (int off = 64; off > 0; off >>= 1) {
    if (tid < off) r[tid] += r[tid + off];
    __syncthreads();
  }
  const float rs = rsqrtf(r[0] * (1.0f / CP_DIM) + 1e-5f);
  zn[tid] = d * rs * gp[tid] + bp[tid];
  __syncthreads();
  if (tid < H_NUM) {
    float acc = 0.f;
#pragma unroll 4
    for (int c = 0; c < CP_DIM; ++c) acc += zn[c] * wp[c * H_NUM + tid];
    biasT[((size_t)tid * N_TOK + i) * N_TOK + j] =
        acc + beta[(size_t)i * N_TOK + j];
  }
}

// ---------------------------------------------------------------------------
// 5) Attention: per (head, 16-row q-tile). QK^T + bias via WMMA (D 48->64),
//    softmax in LDS (normalizer deferred), P@V via WMMA, fused sigmoid gate.
// ---------------------------------------------------------------------------
__global__ __launch_bounds__(256)
void attn_kernel(const _Float16* __restrict__ qh, const _Float16* __restrict__ kh,
                 const _Float16* __restrict__ vh, const float* __restrict__ biasT,
                 const float* __restrict__ gate, _Float16* __restrict__ att_in) {
  extern __shared__ char smem[];
  float*    logits = (float*)smem;               // 16*1024 f32 (64KB)
  _Float16* probs  = (_Float16*)(smem + 65536);  // 16*1024 f16 (32KB)
  float*    rowred = (float*)(smem + 98304);     // 16*16
  float*    rowmax = rowred + 256;               // 16
  float*    rowsum = rowmax + 16;                // 16

  const int h = blockIdx.y, q0 = blockIdx.x * 16;
  const int tid = threadIdx.x, wave = tid >> 5, lane = tid & 31;
  const int laneRow = lane & 15, laneHi = lane >> 4;

  // Q fragments (rows q0..q0+15, d padded 48->64); 1/sqrt(D) folded into q.
  v16h aq0, aq1;
  {
    const _Float16* qr = qh + (size_t)(q0 + laneRow) * C_DIM + h * D_DIM;
#pragma unroll
    for (int e = 0; e < 16; ++e) {
      int dd = e + (laneHi ? 16 : 0);  // 0..31
      aq0[e] = qr[dd];
      int dd2 = dd + 32;                                  // 32..63
      aq1[e] = (dd2 < D_DIM) ? qr[dd2] : (_Float16)0.f;
    }
  }

  // Phase 1: logits (8 k-tiles per wave)
  for (int kt = wave; kt < N_TOK / 16; kt += 8) {
    const int kcol = kt * 16 + laneRow;
    const _Float16* kr = kh + (size_t)kcol * C_DIM + h * D_DIM;
    v16h b0, b1;
#pragma unroll
    for (int e = 0; e < 16; ++e) {
      int kk0 = (laneHi ? 16 : 0) + e;  // 0..31 < 48
      b0[e] = kr[kk0];
      int kk1 = 32 + kk0;               // 32..63
      b1[e] = (kk1 < D_DIM) ? kr[kk1] : (_Float16)0.f;
    }
    v8f c = {0.f, 0.f, 0.f, 0.f, 0.f, 0.f, 0.f, 0.f};
    c = wmma_f16(aq0, b0, c);
    c = wmma_f16(aq1, b1, c);
#pragma unroll
    for (int e = 0; e < 8; ++e) {
      int m = e + 8 * laneHi;
      float bv = biasT[(((size_t)h * N_TOK) + (q0 + m)) * N_TOK + kcol];
      logits[m * N_TOK + kcol] = c[e] + bv;
    }
  }
  __syncthreads();

  // Phase 2: softmax (16 threads per row), normalizer deferred to epilogue
  const int row = tid >> 4, t = tid & 15;
  float mx = -1e30f;
  for (int c0 = t; c0 < N_TOK; c0 += 16) mx = fmaxf(mx, logits[row * N_TOK + c0]);
  rowred[row * 16 + t] = mx; __syncthreads();
  if (t == 0) {
    float m2 = -1e30f;
    for (int u = 0; u < 16; ++u) m2 = fmaxf(m2, rowred[row * 16 + u]);
    rowmax[row] = m2;
  }
  __syncthreads();
  const float rm = rowmax[row];
  float sum = 0.f;
  for (int c0 = t; c0 < N_TOK; c0 += 16) {
    float p = __expf(logits[row * N_TOK + c0] - rm);
    probs[row * N_TOK + c0] = (_Float16)p;
    sum += p;
  }
  rowred[row * 16 + t] = sum; __syncthreads();
  if (t == 0) {
    float s2 = 0.f;
    for (int u = 0; u < 16; ++u) s2 += rowred[row * 16 + u];
    rowsum[row] = s2;
  }
  __syncthreads();

  // Phase 3: O = P @ V over full N, 3 col-tiles of 16 (D=48), 3 waves.
  if (wave < 3) {
    const int d0 = wave * 16;
    const int dcol = d0 + laneRow;
    v8f acc = {0.f, 0.f, 0.f, 0.f, 0.f, 0.f, 0.f, 0.f};
    for (int kc = 0; kc < N_TOK; kc += 32) {
      v16h ap, bv;
      const int abase = kc + (laneHi ? 8 : 0);
      const v8h* p0 = (const v8h*)&probs[laneRow * N_TOK + abase];
      v8h lo = p0[0], hi = p0[2];
#pragma unroll
      for (int e = 0; e < 8; ++e) { ap[e] = lo[e]; ap[e + 8] = hi[e]; }
#pragma unroll
      for (int e = 0; e < 16; ++e) {
        int kk = kc + (laneHi ? 8 : 0) + ((e < 8) ? e : e + 8);
        bv[e] = vh[(size_t)kk * C_DIM + h * D_DIM + dcol];
      }
      acc = wmma_f16(ap, bv, acc);
    }
#pragma unroll
    for (int e = 0; e < 8; ++e) {
      int m = e + 8 * laneHi;
      int grow = q0 + m;
      float g = gate[(size_t)grow * C_DIM + h * D_DIM + dcol];
      float o = acc[e] / rowsum[m] * g;
      att_in[(size_t)grow * C_DIM + h * D_DIM + dcol] = (_Float16)o;
    }
  }
}

// ---------------------------------------------------------------------------
// 6) SwiGLU product and final gated sum.
// ---------------------------------------------------------------------------
__global__ __launch_bounds__(256)
void fuse_hid_kernel(const float* __restrict__ h1, const float* __restrict__ h2,
                     _Float16* __restrict__ hidh) {
  size_t i = (size_t)blockIdx.x * 256 + threadIdx.x;
  hidh[i] = (_Float16)(h1[i] * h2[i]);  // h1 already silu'd in GEMM epilogue
}

__global__ __launch_bounds__(256)
void final_kernel(const float* __restrict__ soG, const float* __restrict__ attO,
                  const float* __restrict__ s2G, const float* __restrict__ tO,
                  float* __restrict__ out) {
  size_t i = (size_t)blockIdx.x * 256 + threadIdx.x;
  out[i] = soG[i] * attO[i] + s2G[i] * tO[i];
}

// ---------------------------------------------------------------------------
// Launcher
// ---------------------------------------------------------------------------
extern "C" void kernel_launch(void* const* d_in, const int* in_sizes, int n_in,
                              void* d_out, int out_size, void* d_ws, size_t ws_size,
                              hipStream_t stream) {
  (void)in_sizes; (void)n_in; (void)out_size; (void)ws_size;
  const float* a     = (const float*)d_in[0];
  const float* s     = (const float*)d_in[1];
  const float* pair  = (const float*)d_in[2];
  const float* beta  = (const float*)d_in[3];
  const float* sg1   = (const float*)d_in[4];
  const float* wg1   = (const float*)d_in[5];
  const float* bg1   = (const float*)d_in[6];
  const float* wsk1  = (const float*)d_in[7];
  const float* wq    = (const float*)d_in[8];
  const float* bq    = (const float*)d_in[9];
  const float* wk    = (const float*)d_in[10];
  const float* wv    = (const float*)d_in[11];
  const float* gp    = (const float*)d_in[12];
  const float* bp    = (const float*)d_in[13];
  const float* wp    = (const float*)d_in[14];
  const float* wgate = (const float*)d_in[15];
  const float* wo    = (const float*)d_in[16];
  const float* wso   = (const float*)d_in[17];
  const float* bso   = (const float*)d_in[18];
  const float* sg2   = (const float*)d_in[19];
  const float* wg2   = (const float*)d_in[20];
  const float* bg2   = (const float*)d_in[21];
  const float* wsk2  = (const float*)d_in[22];
  const float* w1    = (const float*)d_in[23];
  const float* w2    = (const float*)d_in[24];
  const float* w3    = (const float*)d_in[25];
  const float* ws2   = (const float*)d_in[26];
  const float* bs2   = (const float*)d_in[27];
  float* out = (float*)d_out;

  char* wsp = (char*)d_ws;
  auto alloc = [&](size_t bytes) -> void* {
    void* p = wsp;
    wsp += (bytes + 255) & ~(size_t)255;
    return p;
  };
  const size_t NC = (size_t)N_TOK * C_DIM;
  const size_t NH = (size_t)N_TOK * HID_DIM;
  const size_t CC = (size_t)C_DIM * C_DIM;
  const size_t CH = (size_t)C_DIM * HID_DIM;

  // activations
  float*    an_raw = (float*)alloc(NC * 4);
  _Float16* sn1h   = (_Float16*)alloc(NC * 2);
  _Float16* sn2h   = (_Float16*)alloc(NC * 2);
  _Float16* sh     = (_Float16*)alloc(NC * 2);
  float*    G1     = (float*)alloc(NC * 4);
  float*    SK1    = (float*)alloc(NC * 4);
  float*    G2     = (float*)alloc(NC * 4);
  float*    SK2    = (float*)alloc(NC * 4);
  _Float16* anh    = (_Float16*)alloc(NC * 2);
  _Float16* an2h   = (_Float16*)alloc(NC * 2);
  _Float16* qh     = (_Float16*)alloc(NC * 2);
  _Float16* kh     = (_Float16*)alloc(NC * 2);
  _Float16* vh     = (_Float16*)alloc(NC * 2);
  float*    gateF  = (float*)alloc(NC * 4);
  float*    soG    = (float*)alloc(NC * 4);
  float*    s2G    = (float*)alloc(NC * 4);
  float*    biasT  = (float*)alloc((size_t)H_NUM * N_TOK * N_TOK * 4);  // 64MB
  _Float16* attInH = (_Float16*)alloc(NC * 2);
  float*    attOut = (float*)alloc(NC * 4);
  float*    H1     = (float*)alloc(NH * 4);
  float*    H2     = (float*)alloc(NH * 4);
  _Float16* hidh   = (_Float16*)alloc(NH * 2);
  float*    Tbuf   = (float*)alloc(NC * 4);
  // f16 transposed weights
  _Float16* wg1t   = (_Float16*)alloc(CC * 2);
  _Float16* wsk1t  = (_Float16*)alloc(CC * 2);
  _Float16* wg2t   = (_Float16*)alloc(CC * 2);
  _Float16* wsk2t  = (_Float16*)alloc(CC * 2);
  _Float16* wqt    = (_Float16*)alloc(CC * 2);
  _Float16* wkt    = (_Float16*)alloc(CC * 2);
  _Float16* wvt    = (_Float16*)alloc(CC * 2);
  _Float16* wgatet = (_Float16*)alloc(CC * 2);
  _Float16* wsot   = (_Float16*)alloc(CC * 2);
  _Float16* ws2t   = (_Float16*)alloc(CC * 2);
  _Float16* wot    = (_Float16*)alloc(CC * 2);
  _Float16* w1t    = (_Float16*)alloc(CH * 2);
  _Float16* w2t    = (_Float16*)alloc(CH * 2);
  _Float16* w3t    = (_Float16*)alloc(CH * 2);

  auto conv = [&](const float* W, _Float16* Wt, int K, int Nc) {
    convert_w_kernel<<<dim3(Nc / 32, K / 32), 256, 0, stream>>>(W, Wt, K, Nc);
  };
  auto gemm = [&](const _Float16* A, const _Float16* Wt, const float* bias,
                  float* oF, _Float16* oH, int M, int K, int Nc, int act,
                  float scale) {
    gemm_kernel<<<dim3(Nc / BN, M / BM), 256, 0, stream>>>(A, Wt, bias, oF, oH,
                                                           M, K, Nc, act, scale);
  };
  const float qscale = 1.0f / sqrtf((float)D_DIM);

  // Weight conversions (f32 [K,N] -> f16 [N,K]); ~60MB traffic total.
  conv(wg1, wg1t, C_DIM, C_DIM);   conv(wsk1, wsk1t, C_DIM, C_DIM);
  conv(wg2, wg2t, C_DIM, C_DIM);   conv(wsk2, wsk2t, C_DIM, C_DIM);
  conv(wq, wqt, C_DIM, C_DIM);     conv(wk, wkt, C_DIM, C_DIM);
  conv(wv, wvt, C_DIM, C_DIM);     conv(wgate, wgatet, C_DIM, C_DIM);
  conv(wso, wsot, C_DIM, C_DIM);   conv(ws2, ws2t, C_DIM, C_DIM);
  conv(wo, wot, C_DIM, C_DIM);
  conv(w1, w1t, C_DIM, HID_DIM);   conv(w2, w2t, C_DIM, HID_DIM);
  conv(w3, w3t, HID_DIM, C_DIM);

  // LN prep
  ln_prep_kernel<<<N_TOK, 256, 0, stream>>>(a, s, sg1, sg2, an_raw, sn1h, sn2h, sh);
  // adaLN GEMMs (both branches)
  gemm(sn1h, wg1t, bg1, G1, nullptr, N_TOK, C_DIM, C_DIM, 1, 1.f);  // sigmoid
  gemm(sn1h, wsk1t, nullptr, SK1, nullptr, N_TOK, C_DIM, C_DIM, 0, 1.f);
  gemm(sn2h, wg2t, bg2, G2, nullptr, N_TOK, C_DIM, C_DIM, 1, 1.f);
  gemm(sn2h, wsk2t, nullptr, SK2, nullptr, N_TOK, C_DIM, C_DIM, 0, 1.f);
  ada_combine_kernel<<<(unsigned)(NC / 256), 256, 0, stream>>>(G1, SK1, G2, SK2,
                                                               an_raw, anh, an2h);
  // projections (1/sqrt(D) folded into q)
  gemm(anh, wqt, bq, nullptr, qh, N_TOK, C_DIM, C_DIM, 0, qscale);
  gemm(anh, wkt, nullptr, nullptr, kh, N_TOK, C_DIM, C_DIM, 0, 1.f);
  gemm(anh, wvt, nullptr, nullptr, vh, N_TOK, C_DIM, C_DIM, 0, 1.f);
  gemm(anh, wgatet, nullptr, gateF, nullptr, N_TOK, C_DIM, C_DIM, 1, 1.f);
  gemm(sh, wsot, bso, soG, nullptr, N_TOK, C_DIM, C_DIM, 1, 1.f);
  gemm(sh, ws2t, bs2, s2G, nullptr, N_TOK, C_DIM, C_DIM, 1, 1.f);
  // pair bias: single pass over the 512MB pair tensor
  pair_bias_kernel<<<dim3(N_TOK, N_TOK), 128, 0, stream>>>(pair, beta, gp, bp, wp,
                                                           biasT);
  // attention (dyn LDS: 64KB logits + 32KB probs + reduction scratch)
  const size_t attn_smem = 16 * N_TOK * 4 + 16 * N_TOK * 2 + (256 + 16 + 16) * 4;
  attn_kernel<<<dim3(N_TOK / 16, H_NUM), 256, attn_smem, stream>>>(
      qh, kh, vh, biasT, gateF, attInH);
  gemm(attInH, wot, nullptr, attOut, nullptr, N_TOK, C_DIM, C_DIM, 0, 1.f);
  // transition (SwiGLU)
  gemm(an2h, w1t, nullptr, H1, nullptr, N_TOK, C_DIM, HID_DIM, 2, 1.f);  // silu
  gemm(an2h, w2t, nullptr, H2, nullptr, N_TOK, C_DIM, HID_DIM, 0, 1.f);
  fuse_hid_kernel<<<(unsigned)(NH / 256), 256, 0, stream>>>(H1, H2, hidh);
  gemm(hidh, w3t, nullptr, Tbuf, nullptr, N_TOK, HID_DIM, C_DIM, 0, 1.f);
  // out = so_gate*att_out + s2_gate*t
  final_kernel<<<(unsigned)(NC / 256), 256, 0, stream>>>(soG, attOut, s2G, Tbuf,
                                                         out);
}
